// MHA_39608188404275
// MI455X (gfx1250) — matrix operand order repaired and checked by reading, not compiled, not run on hardware
//
#include <hip/hip_runtime.h>
#include <hip/hip_bf16.h>

// MHA forward for MI455X (gfx1250, wave32).
// Pipeline: f32->f16 convert (+weight transpose) -> qkv_proj GEMM (x3, WMMA)
//           -> flash attention (WMMA) -> out_proj GEMM (WMMA).
// LDS staging is double-buffered via CDNA5 global_load_async_to_lds_b128,
// overlapped with 16-WMMA bursts using partial s_wait_asynccnt.

typedef __attribute__((ext_vector_type(16))) _Float16 v16h;
typedef __attribute__((ext_vector_type(8)))  _Float16 v8h;
typedef __attribute__((ext_vector_type(4)))  _Float16 v4h;
typedef __attribute__((ext_vector_type(8)))  float    v8f;
typedef __attribute__((ext_vector_type(4)))  float    v4f;

#define B_   2
#define S_   2048
#define DIM_ 1024
#define H_   16
#define HD_  64
#define MTOT (B_ * S_)   // 4096 rows in the projection GEMMs

static __device__ __forceinline__ v16h make_frag(const _Float16* lo, const _Float16* hi) {
  v8h a = *(const v8h*)lo;
  v8h b = *(const v8h*)hi;
  return __builtin_shufflevector(a, b, 0,1,2,3,4,5,6,7,8,9,10,11,12,13,14,15);
}

static __device__ __forceinline__ v8f wmma16(v16h a, v16h b, v8f c) {
  return __builtin_amdgcn_wmma_f32_16x16x32_f16(false, a, false, b, (short)0, c, false, false);
}

// CDNA5 async copy: 16 bytes global -> LDS, tracked on ASYNCcnt (in-order done).
static __device__ __forceinline__ void async_cp16(const _Float16* g, _Float16* l) {
  unsigned lds_off = (unsigned)(uintptr_t)l;  // LDS aperture: addr[31:0] == LDS offset
  asm volatile("global_load_async_to_lds_b128 %0, %1, off"
               :: "v"(lds_off), "v"(g) : "memory");
}
static __device__ __forceinline__ void async_wait0() {
  asm volatile("s_wait_asynccnt 0x0" ::: "memory");
}
static __device__ __forceinline__ void async_wait8() {
  asm volatile("s_wait_asynccnt 0x8" ::: "memory");
}

// ---------------------------------------------------------------------------
// f32 -> f16 elementwise convert (activations). n % 1024 == 0.
// ---------------------------------------------------------------------------
__global__ __launch_bounds__(256) void cvt_f16(const float* __restrict__ in,
                                               _Float16* __restrict__ out, int n) {
  int i = (blockIdx.x * 256 + threadIdx.x) * 4;
  if (i < n) {
    v4f x = *(const v4f*)(in + i);
    v4h h = { (_Float16)x[0], (_Float16)x[1], (_Float16)x[2], (_Float16)x[3] };
    *(v4h*)(out + i) = h;
  }
}

// ---------------------------------------------------------------------------
// Weight convert + transpose: W f32 [K=1024][N=1024] -> Wt f16 [N][K].
// ---------------------------------------------------------------------------
__global__ __launch_bounds__(256) void cvtT_f16(const float* __restrict__ W,
                                                _Float16* __restrict__ Wt) {
  __shared__ float tile[32][33];
  const int x = threadIdx.x & 31;
  const int y = threadIdx.x >> 5;  // 0..7
  const int n0 = blockIdx.x * 32;
  const int k0 = blockIdx.y * 32;
#pragma unroll
  for (int j = 0; j < 4; ++j) {
    int r = y + j * 8;
    tile[r][x] = W[(size_t)(k0 + r) * DIM_ + n0 + x];
  }
  __syncthreads();
#pragma unroll
  for (int j = 0; j < 4; ++j) {
    int r = y + j * 8;
    Wt[(size_t)(n0 + r) * DIM_ + k0 + x] = (_Float16)tile[x][r];
  }
}

// ---------------------------------------------------------------------------
// QKV projection GEMM: X16[4096,1024] f16 @ Wt16[N][K] f16 + bias -> f16 heads.
// Block tile 128x128, K chunk 64 (2 WMMA k-steps), double-buffered LDS.
// 8 waves in 4(M) x 2(N): each wave 32x64 = 16 WMMA per chunk.
// transpose_out==0: Out[(b*H+h)][s][hd]   (Q, K)
// transpose_out==1: Out[(b*H+h)][hd][s]   (V, pre-transposed for attention)
// ---------------------------------------------------------------------------
__global__ __launch_bounds__(256) void proj_gemm(const _Float16* __restrict__ X16,
                                                 const _Float16* __restrict__ Wt16,
                                                 const float* __restrict__ bias,
                                                 _Float16* __restrict__ Out,
                                                 float scale, int transpose_out) {
  __shared__ _Float16 Ash[2][128 * 72];  // [m][k0..63], stride 72 f16 (144 B)
  __shared__ _Float16 Bsh[2][128 * 72];  // [n][k0..63], stride 72

  const int tid = threadIdx.x;
  const int l   = tid & 31;
  const int w   = tid >> 5;
  const int g   = l >> 4;
  const int ln  = l & 15;
  const int wm  = (w & 3) << 5;   // 0,32,64,96
  const int wn  = (w >> 2) << 6;  // 0,64
  const int n0  = blockIdx.x * 128;
  const int m0  = blockIdx.y * 128;

  v8f acc[2][4] = {};

  // 8 async ops per thread per chunk (4 slots x {A,B})
  auto stageAB = [&](int kc, int bufi) {
#pragma unroll
    for (int ii = 0; ii < 4; ++ii) {
      int i = tid + (ii << 8);
      int r = i >> 3, kq = (i & 7) << 3;
      async_cp16(X16 + (size_t)(m0 + r) * DIM_ + kc + kq, &Ash[bufi][r * 72 + kq]);
      async_cp16(Wt16 + (size_t)(n0 + r) * DIM_ + kc + kq, &Bsh[bufi][r * 72 + kq]);
    }
  };
  auto computeChunk = [&](int bufi) {
#pragma unroll
    for (int ks = 0; ks < 2; ++ks) {
      v16h a[2], b[4];
#pragma unroll
      for (int am = 0; am < 2; ++am) {
        const _Float16* ap = &Ash[bufi][(wm + (am << 4) + ln) * 72 + (ks << 5) + (g << 3)];
        a[am] = make_frag(ap, ap + 16);
      }
#pragma unroll
      for (int s = 0; s < 4; ++s) {
        const _Float16* bp = &Bsh[bufi][(wn + (s << 4) + ln) * 72 + (ks << 5) + (g << 3)];
        b[s] = make_frag(bp, bp + 16);
      }
#pragma unroll
      for (int am = 0; am < 2; ++am)
#pragma unroll
        for (int s = 0; s < 4; ++s)
          acc[am][s] = wmma16(a[am], b[s], acc[am][s]);
    }
  };

  stageAB(0, 0);
  int buf = 0;
  for (int kc = 0; kc < DIM_ - 64; kc += 64) {
    stageAB(kc + 64, buf ^ 1);  // prefetch next chunk (async, other buffer)
    async_wait8();              // oldest 8 (current buffer) have landed
    __syncthreads();
    computeChunk(buf);
    __syncthreads();            // buf reusable for staging next iteration
    buf ^= 1;
  }
  async_wait0();
  __syncthreads();
  computeChunk(buf);

  // Epilogue: bias, scale, scatter to head layout
#pragma unroll
  for (int s = 0; s < 4; ++s) {
    int n = n0 + wn + (s << 4) + ln;
    float bb = bias[n];
    int h = n >> 6, hd = n & 63;
#pragma unroll
    for (int am = 0; am < 2; ++am) {
      int mbase = m0 + wm + (am << 4) + (g << 3);  // 8 consecutive rows
      int bi = mbase >> 11, sr0 = mbase & (S_ - 1);
      size_t bh = (size_t)(bi * H_ + h);
      if (transpose_out) {
        // rows are consecutive 's' -> one 16B store
        v8h pk;
#pragma unroll
        for (int v = 0; v < 8; ++v) pk[v] = (_Float16)((acc[am][s][v] + bb) * scale);
        *(v8h*)&Out[(bh * HD_ + hd) * S_ + sr0] = pk;
      } else {
#pragma unroll
        for (int v = 0; v < 8; ++v)
          Out[(bh * S_ + sr0 + v) * HD_ + hd] =
              (_Float16)((acc[am][s][v] + bb) * scale);
      }
    }
  }
}

// ---------------------------------------------------------------------------
// Causal flash attention. grid = (S/64, B*H), block = 128 (4 waves).
// Qh: [bh][s][hd] (pre-scaled by HD^-0.5), Kh: [bh][s][hd], Vt: [bh][hd][s].
// K/V tiles double-buffered via async-to-LDS.
// ---------------------------------------------------------------------------
__global__ __launch_bounds__(128) void attn(const _Float16* __restrict__ Qh,
                                            const _Float16* __restrict__ Kh,
                                            const _Float16* __restrict__ Vt,
                                            _Float16* __restrict__ Aout) {
  __shared__ _Float16 Ksh[2][64 * 72];  // [key][hd], stride 72 f16
  __shared__ _Float16 Vsh[2][64 * 72];  // [hd][key]
  __shared__ _Float16 Psh[64 * 72];     // [m_local][key], per-wave 16-row slabs

  const int tid = threadIdx.x;
  const int l   = tid & 31;
  const int w   = tid >> 5;
  const int g   = l >> 4;
  const int ln  = l & 15;
  const int qb  = blockIdx.x;
  const int bh  = blockIdx.y;
  const int b   = bh >> 4;
  const int h   = bh & 15;
  const int qrow = qb * 64 + w * 16;
  const size_t headbase = (size_t)bh * S_ * HD_;

  const _Float16* qp = Qh + headbase + (size_t)(qrow + ln) * HD_ + (g << 3);
  v16h qf0 = make_frag(qp, qp + 16);
  v16h qf1 = make_frag(qp + 32, qp + 48);

  v8f o[4] = {};
  float mrow[8], lrow[8];
#pragma unroll
  for (int v = 0; v < 8; ++v) { mrow[v] = -1e30f; lrow[v] = 0.0f; }

  // 8 async ops per thread per key block (4 slots x {K,V})
  auto stageKV = [&](int kb, int bufi) {
    const int key0 = kb * 64;
#pragma unroll
    for (int ii = 0; ii < 4; ++ii) {
      int i = tid + (ii << 7);
      int r = i >> 3, kq = (i & 7) << 3;
      async_cp16(Kh + headbase + (size_t)(key0 + r) * HD_ + kq, &Ksh[bufi][r * 72 + kq]);
      async_cp16(Vt + headbase + (size_t)r * S_ + key0 + kq, &Vsh[bufi][r * 72 + kq]);
    }
  };

  stageKV(0, 0);
  int buf = 0;
  for (int kb = 0; kb <= qb; ++kb) {
    const int key0 = kb * 64;
    __syncthreads();  // all waves done reading buf^1 (prev iter) and Psh
    if (kb < qb) {
      stageKV(kb + 1, buf ^ 1);
      async_wait8();  // oldest 8 (current buffer) landed
    } else {
      async_wait0();
    }
    __syncthreads();  // current buffer visible to all waves

    // Scores: 16 queries x 64 keys
    v8f sc[4] = {};
    v16h kb0[4], kb1[4];
#pragma unroll
    for (int s = 0; s < 4; ++s) {
      const _Float16* kp = &Ksh[buf][((s << 4) + ln) * 72 + (g << 3)];
      kb0[s] = make_frag(kp, kp + 16);
      kb1[s] = make_frag(kp + 32, kp + 48);
    }
#pragma unroll
    for (int s = 0; s < 4; ++s) {
      sc[s] = wmma16(qf0, kb0[s], sc[s]);
      sc[s] = wmma16(qf1, kb1[s], sc[s]);
    }

    // Causal mask + row max (D layout: row = v + 8*g, col = s*16 + ln)
    float rmax[8];
#pragma unroll
    for (int v = 0; v < 8; ++v) rmax[v] = -1e30f;
#pragma unroll
    for (int s = 0; s < 4; ++s) {
      int keyg = key0 + (s << 4) + ln;
#pragma unroll
      for (int v = 0; v < 8; ++v) {
        int qg = qrow + v + (g << 3);
        float x = sc[s][v];
        if (keyg > qg) x = -1e30f;
        sc[s][v] = x;
        rmax[v] = fmaxf(rmax[v], x);
      }
    }
#pragma unroll
    for (int v = 0; v < 8; ++v) {
#pragma unroll
      for (int msk = 1; msk < 16; msk <<= 1)
        rmax[v] = fmaxf(rmax[v], __shfl_xor(rmax[v], msk, 32));
    }

    float alpha[8], psum[8];
#pragma unroll
    for (int v = 0; v < 8; ++v) {
      float mn = fmaxf(mrow[v], rmax[v]);
      alpha[v] = __expf(mrow[v] - mn);
      mrow[v]  = mn;
      psum[v]  = 0.0f;
    }
#pragma unroll
    for (int s = 0; s < 4; ++s) {
#pragma unroll
      for (int v = 0; v < 8; ++v) {
        float e = __expf(sc[s][v] - mrow[v]);
        psum[v] += e;
        Psh[(w * 16 + v + (g << 3)) * 72 + (s << 4) + ln] = (_Float16)e;
      }
    }
#pragma unroll
    for (int v = 0; v < 8; ++v) {
#pragma unroll
      for (int msk = 1; msk < 16; msk <<= 1)
        psum[v] += __shfl_xor(psum[v], msk, 32);
      lrow[v] = lrow[v] * alpha[v] + psum[v];
    }
#pragma unroll
    for (int s = 0; s < 4; ++s)
#pragma unroll
      for (int v = 0; v < 8; ++v) o[s][v] *= alpha[v];

    __syncthreads();  // P visible

    const _Float16* pp = &Psh[(w * 16 + ln) * 72 + (g << 3)];
    v16h p0 = make_frag(pp, pp + 16);
    v16h p1 = make_frag(pp + 32, pp + 48);
    v16h vb0[4], vb1[4];
#pragma unroll
    for (int s = 0; s < 4; ++s) {
      const _Float16* vp = &Vsh[buf][((s << 4) + ln) * 72 + (g << 3)];
      vb0[s] = make_frag(vp, vp + 16);
      vb1[s] = make_frag(vp + 32, vp + 48);
    }
#pragma unroll
    for (int s = 0; s < 4; ++s) {
      o[s] = wmma16(p0, vb0[s], o[s]);
      o[s] = wmma16(p1, vb1[s], o[s]);
    }
    buf ^= 1;
  }

  // Normalize and write f16 [B,S,DIM] with head concat
#pragma unroll
  for (int v = 0; v < 8; ++v) {
    float inv = 1.0f / lrow[v];
    int qg = qrow + v + (g << 3);
    size_t rowoff = ((size_t)(b * S_ + qg)) * DIM_ + h * HD_;
#pragma unroll
    for (int s = 0; s < 4; ++s)
      Aout[rowoff + (s << 4) + ln] = (_Float16)(o[s][v] * inv);
  }
}

// ---------------------------------------------------------------------------
// Output projection: Ao[4096,1024] f16 @ Wot[N][K] f16 + bo -> fp32 d_out.
// Same double-buffered 128x128 tiling, K chunk 64.
// ---------------------------------------------------------------------------
__global__ __launch_bounds__(256) void out_gemm(const _Float16* __restrict__ Ain,
                                                const _Float16* __restrict__ Wt16,
                                                const float* __restrict__ bias,
                                                float* __restrict__ Y) {
  __shared__ _Float16 Ash[2][128 * 72];
  __shared__ _Float16 Bsh[2][128 * 72];

  const int tid = threadIdx.x;
  const int l   = tid & 31;
  const int w   = tid >> 5;
  const int g   = l >> 4;
  const int ln  = l & 15;
  const int wm  = (w & 3) << 5;
  const int wn  = (w >> 2) << 6;
  const int n0  = blockIdx.x * 128;
  const int m0  = blockIdx.y * 128;

  v8f acc[2][4] = {};

  auto stageAB = [&](int kc, int bufi) {
#pragma unroll
    for (int ii = 0; ii < 4; ++ii) {
      int i = tid + (ii << 8);
      int r = i >> 3, kq = (i & 7) << 3;
      async_cp16(Ain + (size_t)(m0 + r) * DIM_ + kc + kq, &Ash[bufi][r * 72 + kq]);
      async_cp16(Wt16 + (size_t)(n0 + r) * DIM_ + kc + kq, &Bsh[bufi][r * 72 + kq]);
    }
  };
  auto computeChunk = [&](int bufi) {
#pragma unroll
    for (int ks = 0; ks < 2; ++ks) {
      v16h a[2], b[4];
#pragma unroll
      for (int am = 0; am < 2; ++am) {
        const _Float16* ap = &Ash[bufi][(wm + (am << 4) + ln) * 72 + (ks << 5) + (g << 3)];
        a[am] = make_frag(ap, ap + 16);
      }
#pragma unroll
      for (int s = 0; s < 4; ++s) {
        const _Float16* bp = &Bsh[bufi][(wn + (s << 4) + ln) * 72 + (ks << 5) + (g << 3)];
        b[s] = make_frag(bp, bp + 16);
      }
#pragma unroll
      for (int am = 0; am < 2; ++am)
#pragma unroll
        for (int s = 0; s < 4; ++s)
          acc[am][s] = wmma16(a[am], b[s], acc[am][s]);
    }
  };

  stageAB(0, 0);
  int buf = 0;
  for (int kc = 0; kc < DIM_ - 64; kc += 64) {
    stageAB(kc + 64, buf ^ 1);
    async_wait8();
    __syncthreads();
    computeChunk(buf);
    __syncthreads();
    buf ^= 1;
  }
  async_wait0();
  __syncthreads();
  computeChunk(buf);

#pragma unroll
  for (int s = 0; s < 4; ++s) {
    int n = n0 + wn + (s << 4) + ln;
    float bb = bias[n];
#pragma unroll
    for (int am = 0; am < 2; ++am)
#pragma unroll
      for (int v = 0; v < 8; ++v) {
        int m = m0 + wm + (am << 4) + v + (g << 3);
        Y[(size_t)m * DIM_ + n] = acc[am][s][v] + bb;
      }
  }
}

// ---------------------------------------------------------------------------
extern "C" void kernel_launch(void* const* d_in, const int* in_sizes, int n_in,
                              void* d_out, int out_size, void* d_ws, size_t ws_size,
                              hipStream_t stream) {
  (void)in_sizes; (void)n_in; (void)out_size; (void)ws_size;
  const float* q  = (const float*)d_in[0];
  const float* k  = (const float*)d_in[1];
  const float* v  = (const float*)d_in[2];
  const float* Wq = (const float*)d_in[3];
  const float* bq = (const float*)d_in[4];
  const float* Wk = (const float*)d_in[5];
  const float* bk = (const float*)d_in[6];
  const float* Wv = (const float*)d_in[7];
  const float* bv = (const float*)d_in[8];
  const float* Wo = (const float*)d_in[9];
  const float* bo = (const float*)d_in[10];

  const size_t act  = (size_t)MTOT * DIM_;  // 4M elements
  const size_t wmat = (size_t)DIM_ * DIM_;  // 1M elements
  _Float16* p   = (_Float16*)d_ws;
  _Float16* q16 = p; p += act;
  _Float16* k16 = p; p += act;
  _Float16* v16 = p; p += act;
  _Float16* Wqt = p; p += wmat;
  _Float16* Wkt = p; p += wmat;
  _Float16* Wvt = p; p += wmat;
  _Float16* Wot = p; p += wmat;
  _Float16* Qh  = p; p += act;
  _Float16* Kh  = p; p += act;
  _Float16* Vt  = p; p += act;   // [bh][hd][s]
  _Float16* Ao  = p; p += act;

  const int cvtBlocks = (int)(act / 1024);          // 4 f32 per thread
  cvt_f16<<<cvtBlocks, 256, 0, stream>>>(q, q16, (int)act);
  cvt_f16<<<cvtBlocks, 256, 0, stream>>>(k, k16, (int)act);
  cvt_f16<<<cvtBlocks, 256, 0, stream>>>(v, v16, (int)act);

  dim3 gT(DIM_ / 32, DIM_ / 32);
  cvtT_f16<<<gT, 256, 0, stream>>>(Wq, Wqt);
  cvtT_f16<<<gT, 256, 0, stream>>>(Wk, Wkt);
  cvtT_f16<<<gT, 256, 0, stream>>>(Wv, Wvt);
  cvtT_f16<<<gT, 256, 0, stream>>>(Wo, Wot);

  dim3 gP(DIM_ / 128, MTOT / 128);  // (8, 32)
  proj_gemm<<<gP, 256, 0, stream>>>(q16, Wqt, bq, Qh, 0.125f, 0);  // HD^-0.5
  proj_gemm<<<gP, 256, 0, stream>>>(k16, Wkt, bk, Kh, 1.0f, 0);
  proj_gemm<<<gP, 256, 0, stream>>>(v16, Wvt, bv, Vt, 1.0f, 1);

  attn<<<dim3(S_ / 64, B_ * H_), 128, 0, stream>>>(Qh, Kh, Vt, Ao);

  out_gemm<<<gP, 256, 0, stream>>>(Ao, Wot, bo, (float*)d_out);
}
